// MultiTaskHTMLModel_738734374951
// MI455X (gfx1250) — matrix operand order repaired and checked by reading, not compiled, not run on hardware
//
#include <hip/hip_runtime.h>
#include <hip/hip_bf16.h>

// ---------------------------------------------------------------------------
// MultiTaskHTMLModel for MI455X (gfx1250, wave32, WMMA)
// B=8, C=10, S=512, H=768, WIN=5
// ---------------------------------------------------------------------------

#define Bsz 8
#define NC  10
#define BC  80      // Bsz*NC
#define SS  512
#define HH  768
#define G4  3072    // 4*H
#define H2  1536    // 2*H

typedef __attribute__((ext_vector_type(16))) __bf16 v16bf;
typedef __attribute__((ext_vector_type(8)))  float  v8f;

__device__ __forceinline__ float sigf(float x) { return 1.0f / (1.0f + __expf(-x)); }

__device__ __forceinline__ __bf16 f2bf(float x) {
    union { float f; unsigned u; } v; v.f = x;
    unsigned r = v.u + 0x7FFFu + ((v.u >> 16) & 1u);
    unsigned short h = (unsigned short)(r >> 16);
    union { unsigned short s; __bf16 b; } o; o.s = h;
    return o.b;
}

// ---------------------------------------------------------------------------
// fp32 -> bf16 conversion (weights, activations)
// ---------------------------------------------------------------------------
__global__ __launch_bounds__(256)
void cvt_f32_bf16(const float* __restrict__ src, __bf16* __restrict__ dst, int n) {
    int i = blockIdx.x * 256 + threadIdx.x;
    if (i < n) dst[i] = f2bf(src[i]);
}

// ---------------------------------------------------------------------------
// scores[b,s] = sigmoid(dot(tok[b,s,:], w_tok) + b_tok)   (one wave per row)
// ---------------------------------------------------------------------------
__global__ __launch_bounds__(256)
void scores_kernel(const float* __restrict__ tok, const float* __restrict__ w_tok,
                   const float* __restrict__ b_tok, float* __restrict__ scores) {
    int wave = threadIdx.x >> 5;
    int lane = threadIdx.x & 31;
    int row  = blockIdx.x * 8 + wave;               // 40960 rows total
    const float* p = tok + (size_t)row * HH;
    float acc = 0.0f;
    #pragma unroll 4
    for (int i = lane; i < HH; i += 32) acc += p[i] * w_tok[i];
    #pragma unroll
    for (int off = 16; off > 0; off >>= 1) acc += __shfl_xor(acc, off, 32);
    if (lane == 0) scores[row] = sigf(acc + b_tok[0]);
}

// ---------------------------------------------------------------------------
// windowed = box filter WIN=5, zero padding, divide by 5 always
// ---------------------------------------------------------------------------
__global__ __launch_bounds__(256)
void window_kernel(const float* __restrict__ scores, float* __restrict__ win) {
    int idx = blockIdx.x * 256 + threadIdx.x;       // 40960
    int b = idx >> 9, s = idx & 511;
    float acc = 0.0f;
    #pragma unroll
    for (int d = -2; d <= 2; ++d) {
        int ss = s + d;
        if (ss >= 0 && ss < SS) acc += scores[(b << 9) + ss];
    }
    win[idx] = acc * 0.2f;
}

// ---------------------------------------------------------------------------
// weighted[bc,h] = sum_s tok[bc,s,h] * win[bc,s]
// ---------------------------------------------------------------------------
__global__ __launch_bounds__(768)
void weighted_kernel(const float* __restrict__ tok, const float* __restrict__ win,
                     float* __restrict__ weighted) {
    int bc = blockIdx.x, h = threadIdx.x;
    __shared__ float wsm[SS];
    if (threadIdx.x < SS) wsm[threadIdx.x] = win[bc * SS + threadIdx.x];
    __syncthreads();
    const float* p = tok + (size_t)bc * SS * HH + h;
    float acc = 0.0f;
    for (int s = 0; s < SS; ++s) {
        if ((s & 15) == 0 && s + 16 < SS) __builtin_prefetch(p + (size_t)(s + 16) * HH, 0, 1);
        acc += p[(size_t)s * HH] * wsm[s];
    }
    weighted[bc * HH + h] = acc;
}

// ---------------------------------------------------------------------------
// bf16 WMMA GEMM:  C[m,n] = sum_k A[m,k]*W[n,k] + bias[n]
// A: (>= tiles*16) x K bf16 row-major (padded rows pre-zeroed), lda = K
// W: N x K bf16 row-major (weight matrix, used as B^T)
// One wave (32 threads) per 16x16 output tile; K multiple of 32.
// ---------------------------------------------------------------------------
__global__ __launch_bounds__(32)
void gemm_bf16_wmma(const __bf16* __restrict__ A, int lda,
                    const __bf16* __restrict__ W, int ldb,
                    const float* __restrict__ bias,
                    float* __restrict__ C, int ldo,
                    int Mvalid, int K, int ntilesN) {
    int tile = blockIdx.x;
    int tm = tile / ntilesN, tn = tile % ntilesN;
    int lane = threadIdx.x & 31;
    int m    = lane & 15;
    int half = lane >> 4;

    // A fragment: lane group 0-15 holds K 0..7 (v0-3) and 16..23 (v4-7);
    //             lane group 16-31 holds K 8..15 and 24..31.
    const __bf16* ap = A + (size_t)(tm * 16 + m) * lda + half * 8;
    // B fragment (32x16): same K layout as A (lanes 0-15 K 0..7,16..23; lanes 16-31 K 8..15,24..31).
    const __bf16* bp = W + (size_t)(tn * 16 + m) * ldb + half * 8;

    v8f acc = {};
    for (int k0 = 0; k0 < K; k0 += 32) {
        v16bf af, bf_;
        ((uint4*)&af)[0]  = *(const uint4*)(ap);
        ((uint4*)&af)[1]  = *(const uint4*)(ap + 16);
        ((uint4*)&bf_)[0] = *(const uint4*)(bp);
        ((uint4*)&bf_)[1] = *(const uint4*)(bp + 16);
        acc = __builtin_amdgcn_wmma_f32_16x16x32_bf16(
            false, af, false, bf_, (short)0, acc, false, false);
        ap += 32; bp += 32;
    }

    int col = tn * 16 + (lane & 15);
    float bv = bias ? bias[col] : 0.0f;
    #pragma unroll
    for (int r = 0; r < 8; ++r) {
        int row = tm * 16 + r + 8 * half;   // C/D layout: lanes 16-31 are rows M+8
        if (row < Mvalid) C[(size_t)row * ldo + col] = acc[r] + bv;
    }
}

// ---------------------------------------------------------------------------
// chunk extraction + zero-padded bf16 staging for LSTM GEMMs
// ---------------------------------------------------------------------------
__global__ __launch_bounds__(256)
void prep_chunks(const float* __restrict__ tok, __bf16* __restrict__ chunk_bf,
                 __bf16* __restrict__ chunk9_bf, __bf16* __restrict__ h_bf) {
    int idx = blockIdx.x * 256 + threadIdx.x;       // 80*768
    int bc = idx / HH, h = idx % HH;
    chunk_bf[idx] = f2bf(tok[(size_t)bc * SS * HH + h]);
    if (idx < 16 * HH) {
        int b = idx / HH;
        float v = (b < Bsz) ? tok[(size_t)(b * NC + 9) * SS * HH + (idx % HH)] : 0.0f;
        chunk9_bf[idx] = f2bf(v);
        h_bf[idx] = f2bf(0.0f);                     // pad rows stay zero forever
    }
}

// ---------------------------------------------------------------------------
// mean over chunks -> bf16 staging (16 rows, 8 valid)
// ---------------------------------------------------------------------------
__global__ __launch_bounds__(256)
void mean_kernel(const float* __restrict__ summ1, __bf16* __restrict__ mean_bf) {
    int idx = blockIdx.x * 256 + threadIdx.x;       // 16*768
    int b = idx / HH, h = idx % HH;
    float acc = 0.0f;
    if (b < Bsz) {
        #pragma unroll
        for (int c = 0; c < NC; ++c) acc += summ1[(size_t)(b * NC + c) * HH + h];
        acc *= (1.0f / NC);
    }
    mean_bf[idx] = f2bf(acc);
}

// ---------------------------------------------------------------------------
// LSTM step with zero incoming state (forward t=0 and backward last-step)
// g = xs[row(b)] + b_hh ; c = sig(i)*tanh(g) ; h = sig(o)*tanh(c)
// ---------------------------------------------------------------------------
__global__ __launch_bounds__(256)
void lstm_step_zero(const float* __restrict__ xs, int rowStride, int rowMul, int rowOff,
                    const float* __restrict__ b_hh,
                    float* __restrict__ hbuf, float* __restrict__ cbuf,
                    __bf16* __restrict__ h_bf) {
    int idx = blockIdx.x * 256 + threadIdx.x;       // 8*768
    int b = idx / HH, j = idx % HH;
    const float* g = xs + (size_t)(b * rowMul + rowOff) * rowStride;
    float gi = g[j]            + b_hh[j];
    float gg = g[2 * HH + j]   + b_hh[2 * HH + j];
    float go = g[3 * HH + j]   + b_hh[3 * HH + j];
    float c  = sigf(gi) * tanhf(gg);                // f-gate * 0 dropped
    float h  = sigf(go) * tanhf(c);
    if (cbuf) cbuf[idx] = c;
    hbuf[idx] = h;
    if (h_bf) h_bf[b * HH + j] = f2bf(h);
}

// ---------------------------------------------------------------------------
// LSTM recurrent step t>=1:  g = xs[b*NC+t] + gmat[b] + b_hh
// ---------------------------------------------------------------------------
__global__ __launch_bounds__(256)
void lstm_step(const float* __restrict__ xs, const float* __restrict__ gmat,
               const float* __restrict__ b_hh,
               float* __restrict__ hbuf, float* __restrict__ cbuf,
               __bf16* __restrict__ h_bf, int t) {
    int idx = blockIdx.x * 256 + threadIdx.x;       // 8*768
    int b = idx / HH, j = idx % HH;
    const float* x  = xs   + (size_t)(b * NC + t) * G4;
    const float* gm = gmat + (size_t)b * G4;
    float gi = x[j]          + gm[j]          + b_hh[j];
    float gf = x[HH + j]     + gm[HH + j]     + b_hh[HH + j];
    float gg = x[2 * HH + j] + gm[2 * HH + j] + b_hh[2 * HH + j];
    float go = x[3 * HH + j] + gm[3 * HH + j] + b_hh[3 * HH + j];
    float c  = sigf(gf) * cbuf[idx] + sigf(gi) * tanhf(gg);
    float h  = sigf(go) * tanhf(c);
    cbuf[idx] = c;
    hbuf[idx] = h;
    h_bf[b * HH + j] = f2bf(h);
}

// ---------------------------------------------------------------------------
// class_probs[b] = sigmoid(dot(hf|hb + summ, w_cls) + b_cls)   (wave per b)
// ---------------------------------------------------------------------------
__global__ __launch_bounds__(256)
void final_kernel(const float* __restrict__ hf, const float* __restrict__ hb,
                  const float* __restrict__ summ, const float* __restrict__ w_cls,
                  const float* __restrict__ b_cls, float* __restrict__ probs) {
    int b = threadIdx.x >> 5;                       // 8 waves, one per batch row
    int lane = threadIdx.x & 31;
    float acc = 0.0f;
    for (int j = lane; j < H2; j += 32) {
        float g = (j < HH) ? hf[b * HH + j] : hb[b * HH + (j - HH)];
        acc += (g + summ[b * H2 + j]) * w_cls[j];
    }
    #pragma unroll
    for (int off = 16; off > 0; off >>= 1) acc += __shfl_xor(acc, off, 32);
    if (lane == 0) probs[b] = sigf(acc + b_cls[0]);
}

// ---------------------------------------------------------------------------
extern "C" void kernel_launch(void* const* d_in, const int* in_sizes, int n_in,
                              void* d_out, int out_size, void* d_ws, size_t ws_size,
                              hipStream_t stream) {
    (void)in_sizes; (void)n_in; (void)out_size; (void)ws_size;

    const float* tok    = (const float*)d_in[0];
    const float* w_tok  = (const float*)d_in[1];
    const float* b_tok  = (const float*)d_in[2];
    const float* w_sum  = (const float*)d_in[3];
    const float* b_sum  = (const float*)d_in[4];
    const float* w_proj = (const float*)d_in[5];
    const float* b_proj = (const float*)d_in[6];
    const float* w_ih_f = (const float*)d_in[7];
    const float* w_hh_f = (const float*)d_in[8];
    const float* b_ih_f = (const float*)d_in[9];
    const float* b_hh_f = (const float*)d_in[10];
    const float* w_ih_b = (const float*)d_in[11];
    const float* b_ih_b = (const float*)d_in[13];
    const float* b_hh_b = (const float*)d_in[14];
    const float* w_cls  = (const float*)d_in[15];
    const float* b_cls  = (const float*)d_in[16];

    float* probs    = (float*)d_out;                       // 8
    float* win_out  = (float*)d_out + 8;                   // 80*512
    float* summ_out = (float*)d_out + 8 + BC * SS;         // 8*1536

    // ---- workspace carve-up (256B aligned) ----
    char*  base = (char*)d_ws;
    size_t off  = 0;
    auto alloc = [&](size_t bytes) -> char* {
        char* p = base + off;
        off = (off + bytes + 255) & ~(size_t)255;
        return p;
    };
    __bf16* wsum_bf   = (__bf16*)alloc((size_t)HH * HH * 2);
    __bf16* wproj_bf  = (__bf16*)alloc((size_t)H2 * HH * 2);
    __bf16* wihf_bf   = (__bf16*)alloc((size_t)G4 * HH * 2);
    __bf16* wihb_bf   = (__bf16*)alloc((size_t)G4 * HH * 2);
    __bf16* whhf_bf   = (__bf16*)alloc((size_t)G4 * HH * 2);
    __bf16* chunk_bf  = (__bf16*)alloc((size_t)BC * HH * 2);
    __bf16* chunk9_bf = (__bf16*)alloc((size_t)16 * HH * 2);
    __bf16* wght_bf   = (__bf16*)alloc((size_t)BC * HH * 2);
    __bf16* mean_bf   = (__bf16*)alloc((size_t)16 * HH * 2);
    __bf16* h_bf      = (__bf16*)alloc((size_t)16 * HH * 2);
    float*  scores    = (float*)alloc((size_t)BC * SS * 4);
    float*  weighted  = (float*)alloc((size_t)BC * HH * 4);
    float*  summ1     = (float*)alloc((size_t)BC * HH * 4);
    float*  xs_f      = (float*)alloc((size_t)BC * G4 * 4);
    float*  xs_b9     = (float*)alloc((size_t)16 * G4 * 4);
    float*  gmat      = (float*)alloc((size_t)16 * G4 * 4);
    float*  h_f       = (float*)alloc((size_t)Bsz * HH * 4);
    float*  c_f       = (float*)alloc((size_t)Bsz * HH * 4);
    float*  h_b       = (float*)alloc((size_t)Bsz * HH * 4);

    auto cvt = [&](const float* s, __bf16* d, int n) {
        cvt_f32_bf16<<<(n + 255) / 256, 256, 0, stream>>>(s, d, n);
    };

    // ---- weight conversions to bf16 ----
    cvt(w_sum,  wsum_bf,  HH * HH);
    cvt(w_proj, wproj_bf, H2 * HH);
    cvt(w_ih_f, wihf_bf,  G4 * HH);
    cvt(w_ih_b, wihb_bf,  G4 * HH);
    cvt(w_hh_f, whhf_bf,  G4 * HH);

    // ---- streaming passes over token_embeddings ----
    scores_kernel  <<<BC * SS / 8, 256, 0, stream>>>(tok, w_tok, b_tok, scores);
    window_kernel  <<<BC * SS / 256, 256, 0, stream>>>(scores, win_out);
    weighted_kernel<<<BC, HH, 0, stream>>>(tok, win_out, weighted);
    cvt(weighted, wght_bf, BC * HH);

    // ---- summary branch: (80x768)@(768x768)^T, chunk-mean, proj GEMM ----
    gemm_bf16_wmma<<<(BC / 16) * (HH / 16), 32, 0, stream>>>(
        wght_bf, HH, wsum_bf, HH, b_sum, summ1, HH, BC, HH, HH / 16);
    mean_kernel<<<16 * HH / 256, 256, 0, stream>>>(summ1, mean_bf);
    gemm_bf16_wmma<<<1 * (H2 / 16), 32, 0, stream>>>(
        mean_bf, HH, wproj_bf, HH, b_proj, summ_out, H2, Bsz, HH, H2 / 16);

    // ---- LSTM input transforms ----
    prep_chunks<<<BC * HH / 256, 256, 0, stream>>>(tok, chunk_bf, chunk9_bf, h_bf);
    gemm_bf16_wmma<<<(BC / 16) * (G4 / 16), 32, 0, stream>>>(
        chunk_bf, HH, wihf_bf, HH, b_ih_f, xs_f, G4, BC, HH, G4 / 16);
    gemm_bf16_wmma<<<1 * (G4 / 16), 32, 0, stream>>>(
        chunk9_bf, HH, wihb_bf, HH, b_ih_b, xs_b9, G4, Bsz, HH, G4 / 16);

    // ---- forward LSTM: step 0 (zero state), then 9 recurrent steps ----
    int ew = Bsz * HH / 256;   // 24 blocks
    lstm_step_zero<<<ew, 256, 0, stream>>>(xs_f, G4, NC, 0, b_hh_f, h_f, c_f, h_bf);
    for (int t = 1; t < NC; ++t) {
        gemm_bf16_wmma<<<1 * (G4 / 16), 32, 0, stream>>>(
            h_bf, HH, whhf_bf, HH, nullptr, gmat, G4, Bsz, HH, G4 / 16);
        lstm_step<<<ew, 256, 0, stream>>>(xs_f, gmat, b_hh_f, h_f, c_f, h_bf, t);
    }

    // ---- backward LSTM: only the t=9 (first backward) step is observed ----
    lstm_step_zero<<<ew, 256, 0, stream>>>(xs_b9, G4, 1, 0, b_hh_b, h_b, nullptr, nullptr);

    // ---- classifier head ----
    final_kernel<<<1, Bsz * 32, 0, stream>>>(h_f, h_b, summ_out, w_cls, b_cls, probs);
}
